// BasicStage_36661840839103
// MI455X (gfx1250) — matrix operand order repaired and checked by reading, not compiled, not run on hardware
//
#include <hip/hip_runtime.h>
#include <hip/hip_bf16.h>

typedef __attribute__((ext_vector_type(16))) _Float16 v16h;
typedef __attribute__((ext_vector_type(8)))  float    v8f;
typedef __attribute__((ext_vector_type(4)))  float    f4;

#define NB 4
#define NC 31
#define NH 64
#define NW 64
#define NTOT (NB*NC*NH*NW)   // 507904
#define NTOT4 (NTOT/4)       // 126976

static __device__ inline v8f v8f_zero() {
  v8f z;
#pragma unroll
  for (int r = 0; r < 8; ++r) z[r] = 0.f;
  return z;
}

static __device__ inline f4 splat4(float x) {
  f4 r;
#pragma unroll
  for (int u = 0; u < 4; ++u) r[u] = x;
  return r;
}

static __device__ inline f4 vmax4(f4 a, f4 b) {
  f4 r;
#pragma unroll
  for (int u = 0; u < 4; ++u) r[u] = fmaxf(a[u], b[u]);
  return r;
}

static __device__ inline f4 vmin4(f4 a, f4 b) {
  f4 r;
#pragma unroll
  for (int u = 0; u < 4; ++u) r[u] = fminf(a[u], b[u]);
  return r;
}

static __device__ inline float softsign(float d, float thr) {
  float a = fmaxf(fabsf(d) - thr, 0.f);
  float sg = (d > 0.f) ? 1.f : ((d < 0.f) ? -1.f : 0.f);
  return a * sg;
}

// ---------------------------------------------------------------------------
// E1 (vectorized b128): R_next and Rq_sim
// ---------------------------------------------------------------------------
__global__ void k_e1(const float* __restrict__ L, const float* __restrict__ M,
                     const float* __restrict__ U_M, const float* __restrict__ Rq,
                     const float* __restrict__ U_Rq, const float* __restrict__ Rqplus,
                     const float* __restrict__ U_Rqplus, const float* __restrict__ al,
                     float* __restrict__ R_next, float* __restrict__ Rq_sim) {
  int i = blockIdx.x * blockDim.x + threadIdx.x;
  if (i >= NTOT4) return;
  float a0 = al[0], a1 = al[1], a4 = al[4];
#define LD4(p) (((const f4*)(p))[i])
  f4 l = LD4(L), m = LD4(M), um = LD4(U_M);
  f4 urq = LD4(U_Rq);
  f4 inv_L = 1.f / (a0 * l * l + a1);
  f4 rn = inv_L * (l * (a0 * m + um) + a1 * LD4(Rq) + urq);
  ((f4*)R_next)[i] = rn;
  ((f4*)Rq_sim)[i] = (a1 * rn + a4 * LD4(Rqplus) + LD4(U_Rqplus) - urq) / (a1 + a4);
#undef LD4
}

// ---------------------------------------------------------------------------
// Embeddings: unshuffle(Rq_sim, S) -> 1x1 convs g/theta/phi   [N, e, hw]
// ---------------------------------------------------------------------------
__global__ void k_embed(const float* __restrict__ x,
                        const float* __restrict__ gw, const float* __restrict__ gb,
                        const float* __restrict__ tw, const float* __restrict__ tb,
                        const float* __restrict__ pw, const float* __restrict__ pb,
                        float* __restrict__ g, float* __restrict__ th, float* __restrict__ ph,
                        int S, int e, int hw, int wdim) {
  int idx = blockIdx.x * blockDim.x + threadIdx.x;
  int total = NB * e * hw;
  if (idx >= total) return;
  int i = idx % hw;
  int tmp = idx / hw;
  int eo = tmp % e;
  int n = tmp / e;
  int hh = i / wdim, ww = i % wdim;
  int cin = NC * S * S;
  float ag = gb[eo], at = tb[eo], ap = pb[eo];
  for (int cu = 0; cu < cin; ++cu) {
    int s12 = cu / NC;
    int c = cu % NC;
    int s1 = s12 / S, s2 = s12 % S;
    float xv = x[((n * NC + c) * NH + hh * S + s1) * NW + ww * S + s2];
    ag += gw[eo * cin + cu] * xv;
    at += tw[eo * cin + cu] * xv;
    ap += pw[eo * cin + cu] * xv;
  }
  g[idx] = ag;
  th[idx] = at;
  ph[idx] = ap;
}

// ---------------------------------------------------------------------------
// Pass A: per-column (j) softmax stats of f = theta^T phi, via WMMA f-tiles.
// ---------------------------------------------------------------------------
__global__ void k_colstats(const float* __restrict__ th, const float* __restrict__ ph,
                           float* __restrict__ cm, float* __restrict__ cs,
                           int e, int hw) {
  int lane = threadIdx.x;
  int lm = lane & 15;
  int hi = lane >> 4;
  int n = blockIdx.y;
  int jt = blockIdx.x * 16;
  int jc = jt + lm;

  v16h b;  // B: phi, K = e (zero padded to 32), N = j
#pragma unroll
  for (int h = 0; h < 16; ++h) {
    int k = h + 16 * hi;
    float v = (k < e) ? ph[(n * e + k) * hw + jc] : 0.f;
    b[h] = (_Float16)v;
  }

  float m = -3.0e38f, s = 0.f;
  for (int it = 0; it < hw; it += 16) {
    if (it + 16 < hw)  // prefetch next theta i-tile (uniform branch)
      __builtin_prefetch(&th[(n * e + (lane & 7)) * hw + it + 16], 0, 1);
    v16h a;  // A: theta^T, M = i rows, K = e (padded)
#pragma unroll
    for (int h = 0; h < 16; ++h) {
      int k = h + 16 * hi;
      float v = (k < e) ? th[(n * e + k) * hw + it + lm] : 0.f;
      a[h] = (_Float16)v;
    }
    v8f f = v8f_zero();
    f = __builtin_amdgcn_wmma_f32_16x16x32_f16(false, a, false, b, (short)0, f,
                                               false, false);
    float tm = f[0];
#pragma unroll
    for (int r = 1; r < 8; ++r) tm = fmaxf(tm, f[r]);
    float nm = fmaxf(m, tm);
    float acc = 0.f;
#pragma unroll
    for (int r = 0; r < 8; ++r) acc += __expf(f[r] - nm);
    s = s * __expf(m - nm) + acc;
    m = nm;
  }
  // combine lane L with lane L^16 (same column j, other 8 rows)
  float m2 = __shfl_xor(m, 16, 32);
  float s2 = __shfl_xor(s, 16, 32);
  float M = fmaxf(m, m2);
  float S_ = s * __expf(m - M) + s2 * __expf(m2 - M);
  if (lane < 16) {
    cm[n * hw + jt + lane] = M;
    cs[n * hw + jt + lane] = S_;
  }
}

// g' = g / colsum (vectorized b128; vectors never cross n or j boundaries)
__global__ void k_gdiv(const float* __restrict__ g, const float* __restrict__ cs,
                       float* __restrict__ gd, int e, int hw) {
  int idx = blockIdx.x * blockDim.x + threadIdx.x;
  int total4 = (NB * e * hw) / 4;
  if (idx >= total4) return;
  int hw4 = hw / 4;
  int j4 = idx % hw4;
  int n = idx / (e * hw4);
  f4 gv = ((const f4*)g)[idx];
  f4 cv = ((const f4*)cs)[n * hw4 + j4];
  ((f4*)gd)[idx] = gv / cv;
}

// ---------------------------------------------------------------------------
// Pass B: flash-style  y[e,i] = sum_j exp(f[i,j]-m_j) * g'[e,j]
// f tile via WMMA; exp'd tile staged in LDS (C/D layout -> A layout);
// second WMMA accumulates y with K = j (32 per step).
// ---------------------------------------------------------------------------
__global__ void k_attn(const float* __restrict__ th, const float* __restrict__ ph,
                       const float* __restrict__ gd, const float* __restrict__ cm,
                       float* __restrict__ y, int e, int hw) {
  __shared__ _Float16 wl[16][34];
  int lane = threadIdx.x;
  int lm = lane & 15;
  int hi = lane >> 4;
  int n = blockIdx.y;
  int it = blockIdx.x * 16;

  v16h a;  // theta^T tile for this i-tile (persistent)
#pragma unroll
  for (int h = 0; h < 16; ++h) {
    int k = h + 16 * hi;
    float v = (k < e) ? th[(n * e + k) * hw + it + lm] : 0.f;
    a[h] = (_Float16)v;
  }

  v8f y0 = v8f_zero();
  v8f y1 = v8f_zero();

  for (int j0 = 0; j0 < hw; j0 += 32) {
    if (j0 + 32 < hw) {  // prefetch next j-chunk of phi and g' (uniform branch)
      __builtin_prefetch(&ph[(n * e + (lane & 7)) * hw + j0 + 32 + 16 * hi], 0, 1);
      __builtin_prefetch(&gd[(n * e + (lane & 7)) * hw + j0 + 32 + 16 * hi], 0, 1);
    }
#pragma unroll
    for (int t = 0; t < 2; ++t) {
      int jc = j0 + 16 * t + lm;
      v16h b;
#pragma unroll
      for (int h = 0; h < 16; ++h) {
        int k = h + 16 * hi;
        float v = (k < e) ? ph[(n * e + k) * hw + jc] : 0.f;
        b[h] = (_Float16)v;
      }
      v8f f = v8f_zero();
      f = __builtin_amdgcn_wmma_f32_16x16x32_f16(false, a, false, b, (short)0, f,
                                                 false, false);
      float mj = cm[n * hw + jc];
#pragma unroll
      for (int r = 0; r < 8; ++r)
        wl[r + 8 * hi][16 * t + lm] = (_Float16)__expf(f[r] - mj);
    }
    __syncthreads();

    v16h aw;  // P tile re-read in A layout (M=i, K=j 0..31)
#pragma unroll
    for (int h = 0; h < 16; ++h) {
      int k = h + 16 * hi;
      aw[h] = wl[lm][k];
    }
    {
      v16h bg;  // g' tile: K=j, N=e columns 0..15
#pragma unroll
      for (int h = 0; h < 16; ++h) {
        int k = h + 16 * hi;
        float v = (lm < e) ? gd[(n * e + lm) * hw + j0 + k] : 0.f;
        bg[h] = (_Float16)v;
      }
      y0 = __builtin_amdgcn_wmma_f32_16x16x32_f16(false, aw, false, bg, (short)0,
                                                  y0, false, false);
    }
    if (e > 16) {  // uniform branch (e==32 only)
      v16h bg;
#pragma unroll
      for (int h = 0; h < 16; ++h) {
        int k = h + 16 * hi;
        bg[h] = (_Float16)gd[(n * e + 16 + lm) * hw + j0 + k];
      }
      y1 = __builtin_amdgcn_wmma_f32_16x16x32_f16(false, aw, false, bg, (short)0,
                                                  y1, false, false);
    }
    __syncthreads();
  }

#pragma unroll
  for (int r = 0; r < 8; ++r) {
    int irow = it + r + 8 * hi;
    if (lm < e) y[(n * e + lm) * hw + irow] = y0[r];
  }
  if (e > 16) {
#pragma unroll
    for (int r = 0; r < 8; ++r) {
      int irow = it + r + 8 * hi;
      y[(n * e + 16 + lm) * hw + irow] = y1[r];
    }
  }
}

// ---------------------------------------------------------------------------
// W 1x1 conv + pixel shuffle back -> res[n, coff+c, y, x]
// ---------------------------------------------------------------------------
__global__ void k_wconv(const float* __restrict__ y, const float* __restrict__ Ww,
                        const float* __restrict__ Wb, float* __restrict__ res,
                        int S, int e, int hw, int wdim, int coff) {
  int idx = blockIdx.x * blockDim.x + threadIdx.x;
  if (idx >= NTOT) return;
  int xx = idx % NW;
  int yy = (idx / NW) % NH;
  int c = (idx / (NH * NW)) % NC;
  int n = idx / (NC * NH * NW);
  int s1 = yy % S, s2 = xx % S;
  int hh = yy / S, ww = xx / S;
  int cu = (s1 * S + s2) * NC + c;
  int i = hh * wdim + ww;
  float acc = Wb[cu];
  for (int eo = 0; eo < e; ++eo) acc += Ww[cu * e + eo] * y[(n * e + eo) * hw + i];
  res[((n * 93 + coff + c) * NH + yy) * NW + xx] = acc;
}

// ---------------------------------------------------------------------------
// Final 3x3 conv (93->31, zero-pad SAME) + bias + residual -> Rq_next
// ---------------------------------------------------------------------------
__global__ void k_conv3(const float* __restrict__ res, const float* __restrict__ w,
                        const float* __restrict__ b, const float* __restrict__ rqsim,
                        float* __restrict__ out) {
  int idx = blockIdx.x * blockDim.x + threadIdx.x;
  if (idx >= NTOT) return;
  int xx = idx % NW;
  int yy = (idx / NW) % NH;
  int o = (idx / (NH * NW)) % NC;
  int n = idx / (NC * NH * NW);
  float acc = b[o] + rqsim[idx];
  for (int ic = 0; ic < 93; ++ic) {
    const float* rp = res + ((n * 93 + ic) * NH) * NW;
    const float* wp = w + (o * 93 + ic) * 9;
#pragma unroll
    for (int ky = 0; ky < 3; ++ky) {
      int iy = yy + ky - 1;
      if ((unsigned)iy < (unsigned)NH) {
#pragma unroll
        for (int kx = 0; kx < 3; ++kx) {
          int ix = xx + kx - 1;
          if ((unsigned)ix < (unsigned)NW) acc += wp[ky * 3 + kx] * rp[iy * NW + ix];
        }
      }
    }
  }
  out[idx] = acc;
}

// ---------------------------------------------------------------------------
// E2 (vectorized b128): everything elementwise after Rq_next except TV tail
// ---------------------------------------------------------------------------
__global__ void k_e2(const float* __restrict__ I_, const float* __restrict__ R_,
                     const float* __restrict__ L_, const float* __restrict__ Lr_,
                     const float* __restrict__ Lt_, const float* __restrict__ Lrp_,
                     const float* __restrict__ Ltp_, const float* __restrict__ URq_,
                     const float* __restrict__ URqp_, const float* __restrict__ ULr_,
                     const float* __restrict__ ULt_, const float* __restrict__ ULrp_,
                     const float* __restrict__ ULtp_, const float* __restrict__ UM_,
                     const float* __restrict__ al, const float* __restrict__ Rn,
                     const float* __restrict__ Rqn,
                     float* __restrict__ oRqp, float* __restrict__ oLn,
                     float* __restrict__ oLrn, float* __restrict__ oLrp,
                     float* __restrict__ oMn, float* __restrict__ oURq,
                     float* __restrict__ oURqp, float* __restrict__ oULr,
                     float* __restrict__ oULrp, float* __restrict__ oUM,
                     float* __restrict__ oLtsim) {
  int i = blockIdx.x * blockDim.x + threadIdx.x;
  if (i >= NTOT4) return;
  float a0 = al[0], a1 = al[1], a2 = al[2], a3 = al[3], a4 = al[4], a5 = al[5], a6 = al[6];
#define LD4(p) (((const f4*)(p))[i])
#define ST4(p, v) (((f4*)(p))[i] = (v))
  f4 z = splat4(0.f);
  f4 rqn = LD4(Rqn);
  f4 urqp = LD4(URqp_);
  f4 rqp = vmin4(vmax4(rqn - urqp / a4, z), splat4(1.f));
  ST4(oRqp, rqp);
  ST4(oURqp, urqp + a4 * (rqn - rqp));
  f4 rn = LD4(Rn);
  f4 l = LD4(L_);
  f4 um = LD4(UM_);
  f4 mn = (LD4(I_) + a0 * l * rn - um) / (1.f + a0);
  ST4(oMn, mn);
  ST4(oURq, LD4(URq_) + a1 * (rn - rqn));
  f4 r = LD4(R_);
  f4 inv_R = 1.f / (a0 * r * r + a2 + a3);
  f4 ulr = LD4(ULr_);
  f4 ult = LD4(ULt_);
  f4 ln = inv_R * (r * (a0 * mn + um) + a2 * LD4(Lr_) + a3 * LD4(Lt_) + ulr + ult);
  ST4(oLn, ln);
  f4 ulrp = LD4(ULrp_);
  f4 lrn = (a2 * ln + a5 * LD4(Lrp_) + ulrp - ulr) / (a2 + a5);  // SVD block = identity
  ST4(oLrn, lrn);
  ST4(oLtsim, (a3 * ln + a6 * LD4(Ltp_) + LD4(ULtp_) - ult) / (a3 + a6));
  f4 lrp = vmax4(lrn - ulrp / a5, z);
  ST4(oLrp, lrp);
  ST4(oULrp, ulrp + a5 * (lrn - lrp));
  ST4(oULr, ulr + a2 * (ln - lrn));
  ST4(oUM, um + a0 * (ln * rn - mn));
#undef LD4
#undef ST4
}

// ---------------------------------------------------------------------------
// TV ISTA stencil (faithful to the reference incl. the dsx+dsx "bug") + tails
// ---------------------------------------------------------------------------
__global__ void k_tv(const float* __restrict__ lts, const float* __restrict__ Ln,
                     const float* __restrict__ ULt_, const float* __restrict__ ULtp_,
                     const float* __restrict__ al, const float* __restrict__ rho_p,
                     const float* __restrict__ thr_p, float* __restrict__ oLt,
                     float* __restrict__ oLtp, float* __restrict__ oULtp,
                     float* __restrict__ oULt) {
  int idx = blockIdx.x * blockDim.x + threadIdx.x;
  if (idx >= NTOT) return;
  float rho = rho_p[0], thr = thr_p[0];
  float a3 = al[3], a6 = al[6];
  int xx = idx % NW;
  int yy = (idx / NW) % NH;
  int base = idx - yy * NW - xx;
  const float* X = lts + base;
  float xc = X[yy * NW + xx];
  int ym1 = yy > 0 ? yy - 1 : 0;
  int yp1 = yy < NH - 1 ? yy + 1 : NH - 1;
  int xm1 = xx > 0 ? xx - 1 : 0;
  int xp1 = xx < NW - 1 ? xx + 1 : NW - 1;
  float xu = X[ym1 * NW + xx];
  float xd = X[yp1 * NW + xx];
  float xl = X[yy * NW + xm1];
  float xr = X[yy * NW + xp1];
  float s0 = softsign(xu - xc, thr);  // yy==0 -> xu==xc -> 0, matches edge pad
  float dsx = (yy < NH - 1) ? (softsign(xc - xd, thr) - s0) : 0.f;
  float lapx = 2.f * xc - xu - xd;
  float lapy = 2.f * xc - xl - xr;
  float ltn = xc - rho * (lapx + lapy) + 2.f * rho * dsx;
  oLt[idx] = ltn;
  float ultp = ULtp_[idx];
  float ltp = fmaxf(ltn - ultp / a6, 0.f);
  oLtp[idx] = ltp;
  oULtp[idx] = ultp + a6 * (ltn - ltp);
  oULt[idx] = ULt_[idx] + a3 * (Ln[idx] - ltn);
}

// ---------------------------------------------------------------------------
extern "C" void kernel_launch(void* const* d_in, const int* in_sizes, int n_in,
                              void* d_out, int out_size, void* d_ws, size_t ws_size,
                              hipStream_t stream) {
  (void)in_sizes; (void)n_in; (void)out_size; (void)ws_size;
#define F(k) ((const float*)d_in[k])
  float* out = (float*)d_out;
#define O(k) (out + (size_t)(k) * NTOT)

  float* ws = (float*)d_ws;
  float* rqsim = ws;                      // 507904
  float* ltsim = rqsim + NTOT;            // 507904
  float* thbuf = ltsim + NTOT;            // 131072 (max over scales)
  float* phbuf = thbuf + 131072;
  float* gbuf  = phbuf + 131072;
  float* gdbuf = gbuf + 131072;
  float* ybuf  = gdbuf + 131072;
  float* cmbuf = ybuf + 131072;           // 16384
  float* csbuf = cmbuf + 16384;           // 16384
  float* resbuf = csbuf + 16384;          // 4*93*64*64 = 1523712

  const int gN = (NTOT + 255) / 256;
  const int gN4 = (NTOT4 + 255) / 256;

  // E1: R_next + Rq_sim
  k_e1<<<gN4, 256, 0, stream>>>(F(4), F(9), F(16), F(2), F(10), F(3), F(11), F(43),
                                O(0), rqsim);

  // Multi-nonlocal: scales 1, 2, 4 (params flattened in dict insertion order)
  struct Sc { int S, e, hw, wdim, coff, base; };
  const Sc scs[3] = { {1, 8, 4096, 64, 0, 17},
                      {2, 16, 1024, 32, 31, 25},
                      {4, 32, 256, 16, 62, 33} };
  for (int s = 0; s < 3; ++s) {
    const Sc sc = scs[s];
    const float* gw = F(sc.base + 0); const float* gb = F(sc.base + 1);
    const float* tw = F(sc.base + 2); const float* tb = F(sc.base + 3);
    const float* pw = F(sc.base + 4); const float* pb = F(sc.base + 5);
    const float* Ww = F(sc.base + 6); const float* Wb = F(sc.base + 7);
    int ne = NB * sc.e * sc.hw;
    k_embed<<<(ne + 255) / 256, 256, 0, stream>>>(rqsim, gw, gb, tw, tb, pw, pb,
                                                  gbuf, thbuf, phbuf, sc.S, sc.e,
                                                  sc.hw, sc.wdim);
    k_colstats<<<dim3(sc.hw / 16, NB), 32, 0, stream>>>(thbuf, phbuf, cmbuf, csbuf,
                                                        sc.e, sc.hw);
    k_gdiv<<<(ne / 4 + 255) / 256, 256, 0, stream>>>(gbuf, csbuf, gdbuf, sc.e, sc.hw);
    k_attn<<<dim3(sc.hw / 16, NB), 32, 0, stream>>>(thbuf, phbuf, gdbuf, cmbuf,
                                                    ybuf, sc.e, sc.hw);
    k_wconv<<<gN, 256, 0, stream>>>(ybuf, Ww, Wb, resbuf, sc.S, sc.e, sc.hw,
                                    sc.wdim, sc.coff);
  }

  // 3x3 conv + bias + residual -> Rq_next
  k_conv3<<<gN, 256, 0, stream>>>(resbuf, F(41), F(42), rqsim, O(1));

  // E2: remaining elementwise (writes Lt_sim for TV)
  k_e2<<<gN4, 256, 0, stream>>>(F(0), F(1), F(4), F(5), F(6), F(7), F(8), F(10),
                                F(11), F(12), F(13), F(14), F(15), F(16), F(43),
                                O(0), O(1),
                                O(2), O(3), O(4), O(6), O(8), O(9), O(10), O(11),
                                O(13), O(15), ltsim);

  // TV + tails
  k_tv<<<gN, 256, 0, stream>>>(ltsim, O(3), F(13), F(15), F(43), F(44), F(45),
                               O(5), O(7), O(14), O(12));
#undef F
#undef O
}
